// network_31464930411078
// MI455X (gfx1250) — compile-verified, matrix operand order checked by reading, not run on hardware
//
#include <hip/hip_runtime.h>
#include <hip/hip_bf16.h>

// ---------------------------------------------------------------------------
// Model dims (fixed by the reference)
// ---------------------------------------------------------------------------
#define BB    64      // batch
#define SS    128     // encoder sequence length
#define EMBD  512
#define HH    1024
#define HENC  512     // HH/2
#define TOUT  50
#define NOUT  10
#define NINN  5

typedef __attribute__((ext_vector_type(16))) __bf16 v16bf;
typedef __attribute__((ext_vector_type(8)))  __bf16 v8bf;
typedef __attribute__((ext_vector_type(8)))  float  v8f;
typedef __attribute__((ext_vector_type(4)))  int    v4i_t;

// CDNA5 async global->LDS path (probe via __has_builtin; fallback = ds_store)
#if defined(__has_builtin)
#if __has_builtin(__builtin_amdgcn_global_load_async_to_lds_b128)
#define USE_ASYNC_LDS 1
#endif
#endif
#ifndef USE_ASYNC_LDS
#define USE_ASYNC_LDS 0
#endif

__device__ __forceinline__ void async_wait0()
{
#if USE_ASYNC_LDS
#if __has_builtin(__builtin_amdgcn_s_wait_asynccnt)
    __builtin_amdgcn_s_wait_asynccnt(0);
#else
    asm volatile("s_wait_asynccnt 0x0" ::: "memory");
#endif
#endif
}

// ---------------------------------------------------------------------------
// WMMA GEMM:  C[64 x N] (f32) = A[64 x K] (bf16) * W[N x K]^T (bf16)
// A described as up to 3 column-segments (ptr, row-stride, width); widths and
// K tiles are multiples of 32 so each 32-wide fragment lies in one segment.
// lda == 0 broadcasts one row to all 64 rows.
//
// Block = 256 threads = 8 waves covering the full 64(M) x 64(N) tile:
//   wave -> mtile = wave&3 (16 rows), ncol = wave>>2 (32 cols, 2 accumulators)
// A tile [64 x 32] staged in LDS (double buffered), async when available.
// ---------------------------------------------------------------------------
__global__ __launch_bounds__(256)
void wmma_gemm_bf16(const __bf16* __restrict__ a0, int lda0, int w0,
                    const __bf16* __restrict__ a1, int lda1, int w1,
                    const __bf16* __restrict__ a2, int lda2, int w2,
                    const __bf16* __restrict__ wgt,   // [N, K] bf16 row-major
                    float* __restrict__ C, int N, int K)
{
    __shared__ __bf16 As[2][64 * 32];          // 2 x 4 KB

    const int tid   = threadIdx.x;
    const int lane  = tid & 31;
    const int wave  = tid >> 5;
    const int mtile = wave & 3;
    const int ncol  = wave >> 2;               // 0..1
    const int nb    = blockIdx.x * 64 + ncol * 32;
    const int half  = lane >> 4;
    const int idx   = lane & 15;

    // cooperative A staging: thread t copies 16B (8 bf16): row t/4, col (t&3)*8
    const int srow = tid >> 2;
    const int scol = (tid & 3) * 8;

    auto stage = [&](int k0, int buf) {
        const __bf16* src;
        if (k0 < w0)            src = a0 + (size_t)srow * lda0 + k0;
        else if (k0 < w0 + w1)  src = a1 + (size_t)srow * lda1 + (k0 - w0);
        else                    src = a2 + (size_t)srow * lda2 + (k0 - w0 - w1);
        src += scol;
        __bf16* dst = &As[buf][srow * 32 + scol];
#if USE_ASYNC_LDS
        __builtin_amdgcn_global_load_async_to_lds_b128(
            (__attribute__((address_space(1))) v4i_t*)(void*)src,
            (__attribute__((address_space(3))) v4i_t*)(void*)dst,
            0, 0);
#else
        *(v8bf*)dst = *(const v8bf*)src;
#endif
    };

    const __bf16* brow0 = wgt + (size_t)(nb + idx) * K;        // n-tile 0, row n
    const __bf16* brow1 = brow0 + (size_t)16 * K;              // n-tile 1

    v8f acc0 = {}, acc1 = {};

    stage(0, 0);
    async_wait0();
    __syncthreads();

    int buf = 0;
    for (int k0 = 0; k0 < K; k0 += 32) {
        const int kn = k0 + 32;
        if (kn < K) stage(kn, buf ^ 1);        // overlap next A tile

        // A fragment from LDS: row = mtile*16+idx; K layout per half-wave
        const __bf16* ar = &As[buf][(mtile * 16 + idx) * 32];
        v8bf alo = *(const v8bf*)(ar + 8 * half);
        v8bf ahi = *(const v8bf*)(ar + 16 + 8 * half);
        v16bf av = __builtin_shufflevector(alo, ahi,
                     0,1,2,3,4,5,6,7,8,9,10,11,12,13,14,15);

        // B^T fragments straight from (L2-resident) packed weights
        const __bf16* b0 = brow0 + k0;
        const __bf16* b1 = brow1 + k0;
        __builtin_prefetch((const void*)(b0 + 128), 0, 1);
        __builtin_prefetch((const void*)(b1 + 128), 0, 1);
        v8bf b0lo = *(const v8bf*)(b0 + 8 * half);
        v8bf b0hi = *(const v8bf*)(b0 + 16 + 8 * half);
        v16bf bv0 = __builtin_shufflevector(b0lo, b0hi,
                      0,1,2,3,4,5,6,7,8,9,10,11,12,13,14,15);
        v8bf b1lo = *(const v8bf*)(b1 + 8 * half);
        v8bf b1hi = *(const v8bf*)(b1 + 16 + 8 * half);
        v16bf bv1 = __builtin_shufflevector(b1lo, b1hi,
                      0,1,2,3,4,5,6,7,8,9,10,11,12,13,14,15);

        acc0 = __builtin_amdgcn_wmma_f32_16x16x32_bf16(
                   false, av, false, bv0, (short)0, acc0, false, false);
        acc1 = __builtin_amdgcn_wmma_f32_16x16x32_bf16(
                   false, av, false, bv1, (short)0, acc1, false, false);

        if (kn < K) async_wait0();
        __syncthreads();
        buf ^= 1;
    }

    // C layout: VGPR r of lane l -> C[mtile*16 + 8*(l>>4) + r][ntile*16 + (l&15)]
    float* cp0 = C + (size_t)(mtile * 16 + 8 * half) * N + nb + idx;
#pragma unroll
    for (int r = 0; r < 8; ++r) cp0[(size_t)r * N] = acc0[r];
    float* cp1 = cp0 + 16;
#pragma unroll
    for (int r = 0; r < 8; ++r) cp1[(size_t)r * N] = acc1[r];
}

// ---------------------------------------------------------------------------
// LSTM cell epilogue: gates g[64,4H] (+ biases) -> c (f32, in place),
// h (bf16 for next WMMA step), optional second bf16 copy of h (strided).
// Gate order i,f,g,o.
// ---------------------------------------------------------------------------
__global__ void lstm_cell_act(const float* __restrict__ g,
                              const float* __restrict__ bih,
                              const float* __restrict__ bhh,
                              float* __restrict__ c,
                              __bf16* __restrict__ h,
                              __bf16* __restrict__ h2p, long h2stride,
                              int Hd)
{
    int i = blockIdx.x * blockDim.x + threadIdx.x;
    if (i >= BB * Hd) return;
    int b = i / Hd, j = i % Hd;
    const float* gr = g + (size_t)b * 4 * Hd;
    float gi = gr[j]          + bih[j]          + bhh[j];
    float gf = gr[Hd + j]     + bih[Hd + j]     + bhh[Hd + j];
    float gg = gr[2 * Hd + j] + bih[2 * Hd + j] + bhh[2 * Hd + j];
    float go = gr[3 * Hd + j] + bih[3 * Hd + j] + bhh[3 * Hd + j];
    float si = 1.f / (1.f + __expf(-gi));
    float sf = 1.f / (1.f + __expf(-gf));
    float so = 1.f / (1.f + __expf(-go));
    float cn = sf * c[i] + si * tanhf(gg);
    float hn = so * tanhf(cn);
    c[i] = cn;
    h[i] = (__bf16)hn;
    if (h2p) h2p[(size_t)b * h2stride + j] = (__bf16)hn;
}

// ---------------------------------------------------------------------------
// Attention core: scores = qin . ctx  -> softmax over S -> cvec (bf16)
// One block per batch element; scores staged in LDS.
// ---------------------------------------------------------------------------
__global__ __launch_bounds__(256)
void attn_softmax_ctx(const float* __restrict__ qin,
                      const __bf16* __restrict__ ctx,   // [B,S,H]
                      __bf16* __restrict__ cvec)        // [B,H]
{
    __shared__ float sc[SS];
    const int b = blockIdx.x, t = threadIdx.x;
    const __bf16* cb = ctx + (size_t)b * SS * HH;

    if (t < SS) {
        const float* q = qin + (size_t)b * HH;
        const __bf16* row = cb + (size_t)t * HH;
        float s = 0.f;
        for (int j = 0; j < HH; j += 4) {
            s += q[j]     * (float)row[j];
            s += q[j + 1] * (float)row[j + 1];
            s += q[j + 2] * (float)row[j + 2];
            s += q[j + 3] * (float)row[j + 3];
        }
        sc[t] = s;
    }
    __syncthreads();
    if (t == 0) {
        float m = sc[0];
        for (int i = 1; i < SS; ++i) m = fmaxf(m, sc[i]);
        float sum = 0.f;
        for (int i = 0; i < SS; ++i) { float e = __expf(sc[i] - m); sc[i] = e; sum += e; }
        float inv = 1.f / sum;
        for (int i = 0; i < SS; ++i) sc[i] *= inv;
    }
    __syncthreads();
    // 256 threads x 4 columns = 1024
    int j = t * 4;
    float a0 = 0.f, a1 = 0.f, a2 = 0.f, a3 = 0.f;
    for (int s = 0; s < SS; ++s) {
        float w = sc[s];
        const __bf16* row = cb + (size_t)s * HH + j;
        a0 += w * (float)row[0];
        a1 += w * (float)row[1];
        a2 += w * (float)row[2];
        a3 += w * (float)row[3];
    }
    __bf16* cv = cvec + (size_t)b * HH + j;
    cv[0] = (__bf16)a0; cv[1] = (__bf16)a1; cv[2] = (__bf16)a2; cv[3] = (__bf16)a3;
}

// tanh epilogue of Wout projection: f32 to d_out slot + bf16 feedback copy
__global__ void tanh_store(const float* __restrict__ g,
                           float* __restrict__ outF,
                           __bf16* __restrict__ outB, int n)
{
    int i = blockIdx.x * blockDim.x + threadIdx.x;
    if (i < n) {
        float v = tanhf(g[i]);
        outF[i] = v;
        outB[i] = (__bf16)v;
    }
}

// f32 weight [N,Ksrc] -> bf16 packed at [N,Kdst] column offset kofs
__global__ void pack_w_bf16(const float* __restrict__ W, int Ksrc,
                            __bf16* __restrict__ dst, int Kdst, int kofs, long n)
{
    long i = (long)blockIdx.x * blockDim.x + threadIdx.x;
    if (i >= n) return;
    long r = i / Ksrc, k = i % Ksrc;
    dst[(size_t)r * Kdst + kofs + k] = (__bf16)W[i];
}

// embedding gather -> bf16; ids==nullptr means id = row + idbias (fixed labels)
__global__ void gather_bf16(const int* __restrict__ ids,
                            const float* __restrict__ table, int D,
                            __bf16* __restrict__ out, long n, int idbias)
{
    long i = (long)blockIdx.x * blockDim.x + threadIdx.x;
    if (i >= n) return;
    long r = i / D, j = i % D;
    int id = ids ? ids[r] : (int)(r + idbias);
    out[i] = (__bf16)table[(size_t)id * D + j];
}

__global__ void zero_u32(unsigned* __restrict__ p, int n)
{
    int i = blockIdx.x * blockDim.x + threadIdx.x;
    if (i < n) p[i] = 0u;
}

// ---------------------------------------------------------------------------
// Host orchestration
// ---------------------------------------------------------------------------
static inline void gemm(hipStream_t st,
                        const __bf16* a0, int l0, int w0,
                        const __bf16* a1, int l1, int w1,
                        const __bf16* a2, int l2, int w2,
                        const __bf16* wgt, float* C, int N, int K)
{
    dim3 grid(N / 64);   // each block covers 64(M) x 64(N)
    wmma_gemm_bf16<<<grid, 256, 0, st>>>(a0, l0, w0, a1, l1, w1, a2, l2, w2,
                                         wgt, C, N, K);
}

extern "C" void kernel_launch(void* const* d_in, const int* in_sizes, int n_in,
                              void* d_out, int out_size, void* d_ws, size_t ws_size,
                              hipStream_t stream)
{
    (void)in_sizes; (void)n_in; (void)out_size; (void)ws_size;
    // ---- inputs (setup_inputs dict order) ----
    const int*   eq     = (const int*)  d_in[0];   // [B,S]
    const int*   dp     = (const int*)  d_in[1];   // [B,50]
    const float* emb_W  = (const float*)d_in[2];   // [32000,EMB]
    const float* pemb_W = (const float*)d_in[3];   // [20000,H]
    const float* Wih_f  = (const float*)d_in[4];
    const float* Whh_f  = (const float*)d_in[5];
    const float* bih_f  = (const float*)d_in[6];
    const float* bhh_f  = (const float*)d_in[7];
    const float* Wih_b  = (const float*)d_in[8];
    const float* Whh_b  = (const float*)d_in[9];
    const float* bih_b  = (const float*)d_in[10];
    const float* bhh_b  = (const float*)d_in[11];
    const float* Wih_d  = (const float*)d_in[12];
    const float* Whh_d  = (const float*)d_in[13];
    const float* bih_d  = (const float*)d_in[14];
    const float* bhh_d  = (const float*)d_in[15];
    const float* Win    = (const float*)d_in[16];
    const float* Wout   = (const float*)d_in[17];
    float* out = (float*)d_out;                    // [50,B,H]

    // ---- workspace carve-out (bytes, 256-aligned) ----
    char* base = (char*)d_ws;
    size_t off = 0;
    auto carve = [&](size_t bytes) -> char* {
        char* p = base + off;
        off = (off + bytes + 255) & ~(size_t)255;
        return p;
    };
    __bf16* encWf  = (__bf16*)carve((size_t)4*HENC * (EMBD+HENC) * 2);   // [2048,1024]
    __bf16* encWb  = (__bf16*)carve((size_t)4*HENC * (EMBD+HENC) * 2);
    __bf16* decW   = (__bf16*)carve((size_t)4*HH   * (3*HH)      * 2);   // [4096,3072]
    __bf16* winW   = (__bf16*)carve((size_t)HH * HH * 2);
    __bf16* woutW  = (__bf16*)carve((size_t)HH * 2*HH * 2);
    __bf16* eq_emb = (__bf16*)carve((size_t)BB*SS*EMBD * 2);             // [B,S,EMB]
    __bf16* eq_out = (__bf16*)carve((size_t)BB*SS*HH * 2);               // [B,S,H]
    __bf16* dp_bf  = (__bf16*)carve((size_t)BB*TOUT*HH * 2);             // [B,50,H]
    __bf16* lab_bf = (__bf16*)carve((size_t)NOUT*HH * 2);                // [10,H]
    __bf16* h_enc  = (__bf16*)carve((size_t)BB*HENC * 2);
    float*  c_enc  = (float*) carve((size_t)BB*HENC * 4);
    float*  g_enc  = (float*) carve((size_t)BB*4*HENC * 4);
    __bf16* h_dec  = (__bf16*)carve((size_t)BB*HH * 2);
    float*  c_dec  = (float*) carve((size_t)BB*HH * 4);
    __bf16* out_bf = (__bf16*)carve((size_t)BB*HH * 2);
    float*  g_dec  = (float*) carve((size_t)BB*4*HH * 4);
    float*  qin    = (float*) carve((size_t)BB*HH * 4);
    __bf16* cvec   = (__bf16*)carve((size_t)BB*HH * 2);
    float*  gout   = (float*) carve((size_t)BB*HH * 4);

    // ---- 1. pack weights to bf16 (concat Wih|Whh along K) ----
    {
        long n;
        n = (long)4*HENC*EMBD;
        pack_w_bf16<<<(n+255)/256,256,0,stream>>>(Wih_f, EMBD, encWf, EMBD+HENC, 0,    n);
        pack_w_bf16<<<(n+255)/256,256,0,stream>>>(Wih_b, EMBD, encWb, EMBD+HENC, 0,    n);
        n = (long)4*HENC*HENC;
        pack_w_bf16<<<(n+255)/256,256,0,stream>>>(Whh_f, HENC, encWf, EMBD+HENC, EMBD, n);
        pack_w_bf16<<<(n+255)/256,256,0,stream>>>(Whh_b, HENC, encWb, EMBD+HENC, EMBD, n);
        n = (long)4*HH*2*HH;
        pack_w_bf16<<<(n+255)/256,256,0,stream>>>(Wih_d, 2*HH, decW, 3*HH, 0,    n);
        n = (long)4*HH*HH;
        pack_w_bf16<<<(n+255)/256,256,0,stream>>>(Whh_d, HH,   decW, 3*HH, 2*HH, n);
        n = (long)HH*HH;
        pack_w_bf16<<<(n+255)/256,256,0,stream>>>(Win,  HH,   winW,  HH,   0, n);
        n = (long)HH*2*HH;
        pack_w_bf16<<<(n+255)/256,256,0,stream>>>(Wout, 2*HH, woutW, 2*HH, 0, n);
    }
    // ---- 2. embedding gathers (f32 table -> bf16) ----
    {
        long n = (long)BB*SS*EMBD;
        gather_bf16<<<(n+255)/256,256,0,stream>>>(eq, emb_W, EMBD, eq_emb, n, 0);
        n = (long)BB*TOUT*HH;
        gather_bf16<<<(n+255)/256,256,0,stream>>>(dp, pemb_W, HH, dp_bf, n, 0);
        n = (long)NOUT*HH;
        gather_bf16<<<(n+255)/256,256,0,stream>>>(nullptr, pemb_W, HH, lab_bf, n, 1);
    }

    const int actBlkE = (BB*HENC + 255)/256, actBlkD = (BB*HH + 255)/256;

    // ---- 3. encoder forward direction ----
    zero_u32<<<(BB*HENC*2/4+255)/256,256,0,stream>>>((unsigned*)h_enc, BB*HENC*2/4);
    zero_u32<<<(BB*HENC+255)/256,256,0,stream>>>((unsigned*)c_enc, BB*HENC);
    for (int s = 0; s < SS; ++s) {
        gemm(stream,
             eq_emb + (size_t)s*EMBD, SS*EMBD, EMBD,   // x_t  [B,EMB]
             h_enc, HENC, HENC,                        // h    [B,Henc]
             h_enc, HENC, 0,
             encWf, g_enc, 4*HENC, EMBD+HENC);
        lstm_cell_act<<<actBlkE,256,0,stream>>>(g_enc, bih_f, bhh_f, c_enc, h_enc,
                                                eq_out + (size_t)s*HH, (long)SS*HH, HENC);
    }
    // ---- 4. encoder backward direction ----
    zero_u32<<<(BB*HENC*2/4+255)/256,256,0,stream>>>((unsigned*)h_enc, BB*HENC*2/4);
    zero_u32<<<(BB*HENC+255)/256,256,0,stream>>>((unsigned*)c_enc, BB*HENC);
    for (int s = SS - 1; s >= 0; --s) {
        gemm(stream,
             eq_emb + (size_t)s*EMBD, SS*EMBD, EMBD,
             h_enc, HENC, HENC,
             h_enc, HENC, 0,
             encWb, g_enc, 4*HENC, EMBD+HENC);
        lstm_cell_act<<<actBlkE,256,0,stream>>>(g_enc, bih_b, bhh_b, c_enc, h_enc,
                                                eq_out + (size_t)s*HH + HENC, (long)SS*HH, HENC);
    }

    // ---- 5. decoder ----
    zero_u32<<<(BB*HH*2/4+255)/256,256,0,stream>>>((unsigned*)h_dec, BB*HH*2/4);
    zero_u32<<<(BB*HH+255)/256,256,0,stream>>>((unsigned*)c_dec, BB*HH);
    for (int o = 0; o < NOUT; ++o) {
        // outer cell: x = [lab (broadcast), h]; A = [lab, h, h]
        gemm(stream,
             lab_bf + (size_t)o*HH, 0, HH,     // stride-0 broadcast row
             h_dec, HH, HH,
             h_dec, HH, HH,
             decW, g_dec, 4*HH, 3*HH);
        // writes h (bf16) and out0 copy into out_bf
        lstm_cell_act<<<actBlkD,256,0,stream>>>(g_dec, bih_d, bhh_d, c_dec, h_dec,
                                                out_bf, (long)HH, HH);
        for (int i = 0; i < NINN; ++i) {
            const int t = o * NINN + i;
            // inner cell: x = [dp_t, out]; A = [dp_t, out, h]
            gemm(stream,
                 dp_bf + (size_t)t*HH, TOUT*HH, HH,
                 out_bf, HH, HH,
                 h_dec, HH, HH,
                 decW, g_dec, 4*HH, 3*HH);
            lstm_cell_act<<<actBlkD,256,0,stream>>>(g_dec, bih_d, bhh_d, c_dec, h_dec,
                                                    nullptr, 0, HH);
            // attention: qin = h @ Win^T
            gemm(stream, h_dec, HH, HH, h_dec, HH, 0, h_dec, HH, 0,
                 winW, qin, HH, HH);
            attn_softmax_ctx<<<BB,256,0,stream>>>(qin, eq_out, cvec);
            // att = tanh([cvec, h] @ Wout^T)
            gemm(stream,
                 cvec, HH, HH,
                 h_dec, HH, HH,
                 h_dec, HH, 0,
                 woutW, gout, HH, 2*HH);
            tanh_store<<<(BB*HH+255)/256,256,0,stream>>>(gout,
                        out + (size_t)t*BB*HH, out_bf, BB*HH);
        }
    }
}
